// CaiQuantLinear_63788854280516
// MI455X (gfx1250) — matrix operand order repaired and checked by reading, compiled, not run on hardware
//
#include <hip/hip_runtime.h>

typedef __attribute__((ext_vector_type(16))) _Float16 v16h;
typedef __attribute__((ext_vector_type(8)))  _Float16 v8h;
typedef __attribute__((ext_vector_type(8)))  float    v8f;
typedef __attribute__((ext_vector_type(4)))  int      v4i;

#if __has_builtin(__builtin_amdgcn_global_load_async_to_lds_b128) && \
    __has_builtin(__builtin_amdgcn_s_wait_asynccnt)
#define USE_ASYNC_LDS 1
// exact parameter types per hipcc diagnostic: arg0 = v4i in global AS, arg1 = v4i in LDS AS
#define G1P(p) ((__attribute__((address_space(1))) v4i*)(p))
#define L3P(p) ((__attribute__((address_space(3))) v4i*)(p))
#endif

#define K_DIM 4096
#define N_DIM 4096
#define M_DIM 8192          // 4 * 2048 rows of x, flattened
#define BM 128
#define BN 128
#define BK 64               // two 16x16x32 K-steps per staged slice
#define ASTR (BK + 8)       // 72 halves = 144B stride (16B aligned, bank-spread)
#define BSTR (BK + 8)

union FragH { v16h v; v8h h[2]; };

__global__ __launch_bounds__(256)
void gptq_wmma_f16_kernel(const _Float16* __restrict__ x,
                          const int*      __restrict__ qweight,
                          const _Float16* __restrict__ scales,
                          const int*      __restrict__ qzeros,
                          const _Float16* __restrict__ bias,
                          float*          __restrict__ out)
{
    __shared__ _Float16 As[2][BM][ASTR];   // double-buffered A: 2 x 18432 B
    __shared__ _Float16 Bs[BN][BSTR];      // dequantized B, transposed [n][k]

    const int tid  = threadIdx.x;
    const int lane = tid & 31;
    const int wid  = tid >> 5;
    const int wm   = wid & 3;          // 4 wave-rows  -> 32 M each
    const int wn   = wid >> 2;         // 2 wave-cols  -> 64 N each
    const int half = lane >> 4;
    const int l16  = lane & 15;

    const int m0 = blockIdx.y * BM;
    const int n0 = blockIdx.x * BN;

    // A staging: thread -> (row, 32-half chunk) = 64B per thread per slice
    const int am = tid >> 1;
    const int ak = (tid & 1) << 5;

    // B staging: thread -> (column, packed-row group of 4)
    const int bc   = tid & 127;
    const int bro  = tid >> 7;          // rows r0+bro+2j, j=0..3
    const int gn_b = n0 + bc;
    const int zcol   = gn_b >> 3;
    const int zshift = (gn_b & 7) << 2;

    const v8f vzero = {0.f, 0.f, 0.f, 0.f, 0.f, 0.f, 0.f, 0.f};
    v8f acc[2][4];
#pragma unroll
    for (int i = 0; i < 2; ++i)
#pragma unroll
        for (int j = 0; j < 4; ++j) acc[i][j] = vzero;

#ifdef USE_ASYNC_LDS
    {   // prologue: async-stage slice 0 into buffer 0 (4 x B128 per thread)
        const _Float16* gp = x + (size_t)(m0 + am) * K_DIM + ak;
#pragma unroll
        for (int c = 0; c < 4; ++c)
            __builtin_amdgcn_global_load_async_to_lds_b128(
                G1P(gp + c * 8), L3P(&As[0][am][ak + c * 8]), 0, 0);
    }
#endif

    for (int k0 = 0; k0 < K_DIM; k0 += BK) {
        const int buf  = (k0 / BK) & 1;
        const int nbuf = buf ^ 1;

        // ---- B-side global fetch into regs (overlaps previous slice's WMMA) ----
        const int g  = k0 >> 7;            // GROUPSIZE=128, BK=64 -> slice within one group
        const int r0 = k0 >> 3;
        const int zword = qzeros[g * (N_DIM >> 3) + zcol];
        const int zp    = ((zword >> zshift) & 0xF) + 1;
        const _Float16 sc = scales[g * N_DIM + gn_b];
        int w[4];
#pragma unroll
        for (int j = 0; j < 4; ++j)
            w[j] = qweight[(size_t)(r0 + bro + 2 * j) * N_DIM + gn_b];
        __builtin_prefetch(&qweight[(size_t)(((r0 + 8) & 511) + bro) * N_DIM + gn_b], 0, 3);

#ifndef USE_ASYNC_LDS
        const _Float16* apc = x + (size_t)(m0 + am) * K_DIM + k0 + ak;
        v8h areg[4];
#pragma unroll
        for (int c = 0; c < 4; ++c) areg[c] = *(const v8h*)(apc + c * 8);
#endif

        __syncthreads();   // all waves finished reading LDS for the previous slice

#ifdef USE_ASYNC_LDS
        {   // async-stage the NEXT slice into the other buffer while we compute this one
            const int kn = (k0 + BK) & (K_DIM - 1);   // wraps on last iter (redundant, harmless)
            const _Float16* gp = x + (size_t)(m0 + am) * K_DIM + kn + ak;
#pragma unroll
            for (int c = 0; c < 4; ++c)
                __builtin_amdgcn_global_load_async_to_lds_b128(
                    G1P(gp + c * 8), L3P(&As[nbuf][am][ak + c * 8]), 0, 0);
        }
#else
#pragma unroll
        for (int c = 0; c < 4; ++c) *(v8h*)&As[buf][am][ak + c * 8] = areg[c];
#endif

        // ---- dequant + stage B (transposed [n][k]) ----
#pragma unroll
        for (int j = 0; j < 4; ++j) {
            v8h bqv;
#pragma unroll
            for (int i = 0; i < 8; ++i)
                bqv[i] = (_Float16)(((w[j] >> (4 * i)) & 0xF) - zp) * sc;
            *(v8h*)&Bs[bc][(bro + 2 * j) * 8] = bqv;   // K = (bro+2j)*8 .. +7
        }

#ifdef USE_ASYNC_LDS
        // async loads complete in order: leaving the 4 newest (next slice)
        // outstanding guarantees the current slice's 4 copies have landed.
        __builtin_amdgcn_s_wait_asynccnt(4);
#endif
        __syncthreads();

        // ---- WMMA: 2 (M) x 4 (N) tiles, two K=32 steps ----
#pragma unroll
        for (int kk = 0; kk < BK; kk += 32) {
            FragH a[2];
#pragma unroll
            for (int mt = 0; mt < 2; ++mt) {
                const _Float16* p = &As[buf][wm * 32 + mt * 16 + l16][kk];
                // ISA A layout: VGPR0-3 K = 8*half+0..7, VGPR4-7 K = 16+8*half+0..7
                a[mt].h[0] = *(const v8h*)(p + half * 8);
                a[mt].h[1] = *(const v8h*)(p + 16 + half * 8);
            }
#pragma unroll
            for (int nt = 0; nt < 4; ++nt) {
                FragH b;
                const _Float16* p = &Bs[wn * 64 + nt * 16 + l16][kk];
                // B layout matches A: VGPR0-3 K = 8*half+0..7, VGPR4-7 K = 16+8*half+0..7
                b.h[0] = *(const v8h*)(p + half * 8);
                b.h[1] = *(const v8h*)(p + 16 + half * 8);
#pragma unroll
                for (int mt = 0; mt < 2; ++mt) {
                    acc[mt][nt] = __builtin_amdgcn_wmma_f32_16x16x32_f16(
                        false, a[mt].v, false, b.v, (short)0, acc[mt][nt], false, false);
                }
            }
        }
    }

    // ---- epilogue: bias add, round through fp16 (reference is fp16), store ----
#pragma unroll
    for (int nt = 0; nt < 4; ++nt) {
        const int gn = n0 + wn * 64 + nt * 16 + l16;
        const float bz = (float)bias[gn];
#pragma unroll
        for (int mt = 0; mt < 2; ++mt) {
            const int mbase = m0 + wm * 32 + mt * 16 + half * 8;
#pragma unroll
            for (int r = 0; r < 8; ++r) {
                _Float16 hv = (_Float16)(acc[mt][nt][r] + bz);
                out[(size_t)(mbase + r) * N_DIM + gn] = (float)hv;
            }
        }
    }
}

extern "C" void kernel_launch(void* const* d_in, const int* in_sizes, int n_in,
                              void* d_out, int out_size, void* d_ws, size_t ws_size,
                              hipStream_t stream) {
    const _Float16* x       = (const _Float16*)d_in[0];
    const int*      qweight = (const int*)d_in[1];
    const _Float16* scales  = (const _Float16*)d_in[2];
    const int*      qzeros  = (const int*)d_in[3];
    // d_in[4] = g_idx: uniform arange(K)//128, computed analytically in-kernel
    const _Float16* bias    = (const _Float16*)d_in[5];
    float* out = (float*)d_out;

    dim3 grid(N_DIM / BN, M_DIM / BM);   // 32 x 64 = 2048 workgroups
    gptq_wmma_f16_kernel<<<grid, 256, 0, stream>>>(x, qweight, scales, qzeros, bias, out);
}
